// xModalKD_50500225466572
// MI455X (gfx1250) — compile-verified, matrix operand order checked by reading, not run on hardware
//
#include <hip/hip_runtime.h>

// ---------------- problem dims ----------------
#define XB    2
#define XN    120000
#define XS    40000
#define XM    30000
#define XV    100000
#define XH    256
#define XW    1024
#define XHID  128
#define XNSC  4
#define XNCLS 20
#define XBN   (XB * XN)   // 240000
#define XBM   (XB * XM)   // 60000

typedef __attribute__((ext_vector_type(16))) _Float16 v16h;
typedef __attribute__((ext_vector_type(8)))  float    v8f;

#define V8F_ZERO {0.f,0.f,0.f,0.f,0.f,0.f,0.f,0.f}

// pointer types for the async global->LDS builtin:
// (v4i addrspace(1)* src, v4i addrspace(3)* dst, imm offset, imm cpol)
typedef int v4i __attribute__((vector_size(4 * sizeof(int))));
typedef __attribute__((address_space(1))) v4i gv4i;
typedef __attribute__((address_space(3))) v4i sv4i;

__device__ __forceinline__ v8f wmma_f32_f16(v16h a, v16h b, v8f c) {
  // D = A(16x32 f16) * B(32x16 f16) + C(16x16 f32)
  return __builtin_amdgcn_wmma_f32_16x16x32_f16(false, a, false, b, (short)0, c, false, false);
}

// Load a 16x32 f16 fragment from row-major storage (ld in halfs).
// A tiles: lane slot = M. B tiles come from pre-transposed weights (N rows x
// K cols), lane slot = N. ISA 7.12.2 K-pair pattern.
__device__ __forceinline__ v16h frag_ld(const _Float16* p, int ld, int lane) {
  const int m = lane & 15;
  const int h = lane >> 4;
  const _Float16* row = p + m * ld + (h << 3);
  v16h f;
#pragma unroll
  for (int r = 0; r < 8; ++r) {
    const int k = ((r & 3) << 1) + ((r >> 2) << 4);
    f[2 * r]     = row[k];
    f[2 * r + 1] = row[k + 1];
  }
  return f;
}

// ---- gfx1250 async global->LDS copy (ASYNCcnt) ----
__device__ __forceinline__ void async_ld_b128(void* lds_dst, const void* gsrc) {
#if __has_builtin(__builtin_amdgcn_global_load_async_to_lds_b128)
  __builtin_amdgcn_global_load_async_to_lds_b128(
      (gv4i*)(uintptr_t)gsrc, (sv4i*)(uint32_t)(uintptr_t)lds_dst, 0, 0);
#else
  asm volatile("global_load_async_to_lds_b128 %0, %1, off"
               :: "v"((uint32_t)(uintptr_t)lds_dst), "v"((uint64_t)(uintptr_t)gsrc)
               : "memory");
#endif
}

__device__ __forceinline__ void async_wait_all() {
#if __has_builtin(__builtin_amdgcn_s_wait_asynccnt)
  __builtin_amdgcn_s_wait_asynccnt(0);
#else
  asm volatile("s_wait_asynccnt 0x0" ::: "memory");
#endif
}

// cooperative async copy, 16B per lane per issue (bytes % 16 == 0)
__device__ __forceinline__ void stage_async(void* lds, const void* g, int bytes, int tid) {
  char* l = (char*)lds;
  const char* s = (const char*)g;
  for (int off = tid * 16; off < bytes; off += 256 * 16)
    async_ld_b128(l + off, s + off);
}

// ---------------- weight prep: f32 (K x 128) -> f16 N-major (128 x K) ----------------
__global__ void transpose_f16_kernel(const float* __restrict__ src, _Float16* __restrict__ dst,
                                     int K) {
  const int idx = blockIdx.x * blockDim.x + threadIdx.x;  // k*128 + n
  if (idx >= K * 128) return;
  const int k = idx >> 7, n = idx & 127;
  dst[n * K + k] = (_Float16)src[idx];
}

// cls_W2 (128 x 20) -> f16 (32 x 128), cols 20..31 zero
__global__ void pad_w2_kernel(const float* __restrict__ src, _Float16* __restrict__ dst) {
  const int idx = blockIdx.x * blockDim.x + threadIdx.x;  // n*128 + k
  if (idx >= 32 * 128) return;
  const int n = idx >> 7, k = idx & 127;
  dst[idx] = (n < XNCLS) ? (_Float16)src[k * XNCLS + n] : (_Float16)0.f;
}

// ---------------- winner-resolution kernels ----------------
__global__ void win_scatter_kernel(const int* __restrict__ idx, int* __restrict__ win,
                                   int count, int per_b, int stride) {
  int t = blockIdx.x * blockDim.x + threadIdx.x;
  if (t >= count) return;
  int b = t / per_b;
  int j = t - b * per_b;
  atomicMax(&win[b * stride + idx[t]], j);
}

__global__ void vox_winner_kernel(const int* __restrict__ coors_inv, int* __restrict__ vwin) {
  int j = blockIdx.x * blockDim.x + threadIdx.x;
  if (j < XBN) atomicMax(&vwin[coors_inv[j]], j);
}

__global__ void gidx_kernel(const int* __restrict__ p2i, const int* __restrict__ coors_inv,
                            const int* __restrict__ vwin, int* __restrict__ gidx) {
  int t = blockIdx.x * blockDim.x + threadIdx.x;
  if (t >= XBM) return;
  int b = t / XM;
  int n = p2i[t];
  gidx[t] = vwin[coors_inv[b * XN + n]];  // always >= 0 (row writes its own voxel)
}

// ---------------- sp = scatter(sample) + scatter(img); writes output_3d ----------------
__global__ void build_sp_kernel(const float* __restrict__ img,    // (B,HID,H,W)
                                const float* __restrict__ sfeat,  // (NSC,B*S,HID)
                                const int*   __restrict__ pts_img,// (B,M,2)
                                const int*   __restrict__ img_w,  // (BN)
                                const int*   __restrict__ smp_w,  // (BN)
                                float*       __restrict__ out3d)  // (BN,512)
{
  const int row = blockIdx.x;          // 0..BN-1
  const int c   = threadIdx.x;         // 0..127
  const int b   = row / XN;

  float pv = 0.f;
  const int mw = img_w[row];
  if (mw >= 0) {
    const int rr = pts_img[((size_t)b * XM + mw) * 2 + 0];
    const int cc = pts_img[((size_t)b * XM + mw) * 2 + 1];
    pv = img[(((size_t)b * XHID + c) * XH + rr) * XW + cc];
  }
  const int sw = smp_w[row];
  float* o = out3d + (size_t)row * (XNSC * XHID);
  if (sw >= 0) {
    const float* sf = sfeat + ((size_t)b * XS + sw) * XHID + c;
#pragma unroll
    for (int i = 0; i < XNSC; ++i)
      o[i * XHID + c] = pv + sf[(size_t)i * XB * XS * XHID];
  } else {
#pragma unroll
    for (int i = 0; i < XNSC; ++i)
      o[i * XHID + c] = pv;
  }
}

// ---------------- fused per-scale MLP (WMMA) ----------------
#define MLP_RPS    128   // rows per strip (8 waves x 16)
#define MLP_STRIPS 2
#define MLP_SMEM_BYTES ((128*128 + 128*256 + 128*128 + MLP_RPS*256 + MLP_RPS*128) * 2 + 3 * 128 * 4)

__global__ __launch_bounds__(256) void fused_mlp_kernel(
    const float*    __restrict__ out3d,    // (BN,512) sp values, f32
    const int*      __restrict__ gidx,     // (BM)
    const float*    __restrict__ img_feat, // (NSC,BM,HID)
    const _Float16* __restrict__ wtl,      // (NSC,128,128) N-major f16
    const _Float16* __restrict__ wt1,      // (NSC,128,256)
    const _Float16* __restrict__ wt2,      // (NSC,128,128)
    const float* __restrict__ lb, const float* __restrict__ f1b, const float* __restrict__ f2b,
    _Float16* __restrict__ seg)            // (BM,512) f16
{
  extern __shared__ char dyn_smem[];
  _Float16* wl  = (_Float16*)dyn_smem;       // 128 x 128 (N-major, K contiguous)
  _Float16* w1  = wl  + 128 * 128;           // 128 x 256
  _Float16* w2  = w1  + 128 * 256;           // 128 x 128
  _Float16* pa2 = w2  + 128 * 128;           // RPS x 256 : [img | fl]
  _Float16* pa1 = pa2 + MLP_RPS * 256;       // RPS x 128 : pts_g, reused for fc(f16)
  float* bl = (float*)(pa1 + MLP_RPS * 128); // 128
  float* b1 = bl + 128;
  float* b2 = b1 + 128;

  const int s    = blockIdx.y;
  const int tid  = threadIdx.x;
  const int lane = tid & 31;
  const int wv   = tid >> 5;                 // 0..7
  const int wg_row0 = blockIdx.x * (MLP_RPS * MLP_STRIPS);

  // --- weights: raw async copy of pre-transposed f16 ---
  stage_async(wl, wtl + (size_t)s * 128 * 128, 128 * 128 * 2, tid);
  stage_async(w1, wt1 + (size_t)s * 128 * 256, 128 * 256 * 2, tid);
  stage_async(w2, wt2 + (size_t)s * 128 * 128, 128 * 128 * 2, tid);
  if (tid < 128) {
    bl[tid] = lb[s * 128 + tid];
    b1[tid] = f1b[s * 128 + tid];
    b2[tid] = f2b[s * 128 + tid];
  }
  async_wait_all();
  __syncthreads();

  for (int st = 0; st < MLP_STRIPS; ++st) {
    const int row0 = wg_row0 + st * MLP_RPS;

    // stage A1 (gathered pts_g rows) and A2 left half (img_feat rows); f32->f16
    for (int idx = tid; idx < MLP_RPS * 128; idx += 256) {
      const int r = idx >> 7, c = idx & 127;
      int rc = row0 + r; if (rc >= XBM) rc = XBM - 1;
      const int g = gidx[rc];
      pa1[r * 128 + c] = (_Float16)out3d[(size_t)g * 512 + s * 128 + c];
      pa2[r * 256 + c] = (_Float16)img_feat[((size_t)s * XBM + rc) * 128 + c];
    }
    __syncthreads();

    const int wr0 = wv * 16;
    const int nn  = lane & 15;
    const int hh  = lane >> 4;

    // GEMM1: fl = relu(A1 @ Wl + bl) -> pa2 right half
#pragma unroll
    for (int nt = 0; nt < 8; ++nt) {
      v8f acc = V8F_ZERO;
#pragma unroll
      for (int kt = 0; kt < 4; ++kt)
        acc = wmma_f32_f16(frag_ld(pa1 + wr0 * 128 + kt * 32, 128, lane),
                           frag_ld(wl + (nt * 16) * 128 + kt * 32, 128, lane), acc);
      const float bb = bl[nt * 16 + nn];
#pragma unroll
      for (int r = 0; r < 8; ++r) {
        float v = acc[r] + bb;
        pa2[(wr0 + r + 8 * hh) * 256 + 128 + nt * 16 + nn] = (_Float16)(v > 0.f ? v : 0.f);
      }
    }
    __syncthreads();

    // GEMM2: fc = A2 @ W1 + b1 ; keep f32 in regs, f16 copy into pa1
    v8f fc[8];
#pragma unroll
    for (int nt = 0; nt < 8; ++nt) {
      v8f acc = V8F_ZERO;
#pragma unroll
      for (int kt = 0; kt < 8; ++kt)
        acc = wmma_f32_f16(frag_ld(pa2 + wr0 * 256 + kt * 32, 256, lane),
                           frag_ld(w1 + (nt * 16) * 256 + kt * 32, 256, lane), acc);
      const float bb = b1[nt * 16 + nn];
#pragma unroll
      for (int r = 0; r < 8; ++r) {
        const float v = acc[r] + bb;
        acc[r] = v;
        pa1[(wr0 + r + 8 * hh) * 128 + nt * 16 + nn] = (_Float16)v;
      }
      fc[nt] = acc;
    }
    __syncthreads();

    // GEMM3: fw = sigmoid(fc @ W2 + b2); fuse = relu(fc * fw) -> seg (f16)
#pragma unroll
    for (int nt = 0; nt < 8; ++nt) {
      v8f acc = V8F_ZERO;
#pragma unroll
      for (int kt = 0; kt < 4; ++kt)
        acc = wmma_f32_f16(frag_ld(pa1 + wr0 * 128 + kt * 32, 128, lane),
                           frag_ld(w2 + (nt * 16) * 128 + kt * 32, 128, lane), acc);
      const float bb = b2[nt * 16 + nn];
#pragma unroll
      for (int r = 0; r < 8; ++r) {
        const int rowg = row0 + wr0 + r + 8 * hh;
        if (rowg < XBM) {
          const float fw = 1.f / (1.f + __expf(-(acc[r] + bb)));
          const float v  = fc[nt][r] * fw;
          seg[(size_t)rowg * 512 + s * 128 + nt * 16 + nn] = (_Float16)(v > 0.f ? v : 0.f);
        }
      }
    }
    __syncthreads();
  }
}

// ---------------- classifier: logits = relu(seg@W1+b1)@W2+b2 ----------------
#define CLS_RPS    128
#define CLS_STRIPS 2
#define CLS_SMEM_BYTES ((128*512 + 32*128 + CLS_RPS*256 + CLS_RPS*128) * 2 + (128 + 32) * 4)

__global__ __launch_bounds__(256) void classifier_kernel(
    const _Float16* __restrict__ seg,  // (BM,512) f16
    const _Float16* __restrict__ cw1,  // (128,512) N-major f16
    const _Float16* __restrict__ cw2,  // (32,128)  N-major f16 (padded)
    const float* __restrict__ b1, const float* __restrict__ b2,
    float* __restrict__ logits)
{
  extern __shared__ char dyn_smem[];
  _Float16* w1t = (_Float16*)dyn_smem;        // 128 x 512
  _Float16* w2t = w1t + 128 * 512;            // 32 x 128
  _Float16* pa  = w2t + 32 * 128;             // RPS x 256 k-chunk of seg
  _Float16* ph  = pa + CLS_RPS * 256;         // RPS x 128 hidden
  float* sb1 = (float*)(ph + CLS_RPS * 128);  // 128
  float* sb2 = sb1 + 128;                     // 32

  const int tid  = threadIdx.x;
  const int lane = tid & 31;
  const int wv   = tid >> 5;
  const int wr0  = wv * 16;
  const int nn   = lane & 15;
  const int hh   = lane >> 4;
  const int wg_row0 = blockIdx.x * (CLS_RPS * CLS_STRIPS);

  stage_async(w1t, cw1, 128 * 512 * 2, tid);
  stage_async(w2t, cw2, 32 * 128 * 2, tid);
  if (tid < 128) sb1[tid] = b1[tid];
  if (tid < 32)  sb2[tid] = (tid < XNCLS) ? b2[tid] : 0.f;
  async_wait_all();
  __syncthreads();

  for (int st = 0; st < CLS_STRIPS; ++st) {
    const int row0 = wg_row0 + st * CLS_RPS;
    v8f acc[8];
#pragma unroll
    for (int nt = 0; nt < 8; ++nt) { v8f z = V8F_ZERO; acc[nt] = z; }

    for (int kk = 0; kk < 2; ++kk) {   // stream K=512 in two 256 chunks
      // raw async copy of the f16 seg chunk (512B per row)
      for (int t = tid; t < CLS_RPS * 32; t += 256) {
        const int r = t >> 5, c16 = t & 31;
        int rc = row0 + r; if (rc >= XBM) rc = XBM - 1;
        async_ld_b128((char*)pa + (size_t)r * 512 + c16 * 16,
                      (const char*)(seg + (size_t)rc * 512 + kk * 256) + c16 * 16);
      }
      async_wait_all();
      __syncthreads();
#pragma unroll
      for (int nt = 0; nt < 8; ++nt)
#pragma unroll
        for (int kt = 0; kt < 8; ++kt)
          acc[nt] = wmma_f32_f16(frag_ld(pa + wr0 * 256 + kt * 32, 256, lane),
                                 frag_ld(w1t + (nt * 16) * 512 + kk * 256 + kt * 32, 512, lane),
                                 acc[nt]);
      __syncthreads();
    }

    // hidden = relu(acc + b1) -> ph (f16)
#pragma unroll
    for (int nt = 0; nt < 8; ++nt) {
      const float bb = sb1[nt * 16 + nn];
#pragma unroll
      for (int r = 0; r < 8; ++r) {
        float v = acc[nt][r] + bb;
        ph[(wr0 + r + 8 * hh) * 128 + nt * 16 + nn] = (_Float16)(v > 0.f ? v : 0.f);
      }
    }
    __syncthreads();

    // logits = hidden @ W2 + b2 (20 cols -> 2 masked n-tiles)
#pragma unroll
    for (int nt = 0; nt < 2; ++nt) {
      v8f a2 = V8F_ZERO;
#pragma unroll
      for (int kt = 0; kt < 4; ++kt)
        a2 = wmma_f32_f16(frag_ld(ph + wr0 * 128 + kt * 32, 128, lane),
                          frag_ld(w2t + (nt * 16) * 128 + kt * 32, 128, lane), a2);
      const int col = nt * 16 + nn;
      const float bb = sb2[col];
#pragma unroll
      for (int r = 0; r < 8; ++r) {
        const int rowg = row0 + wr0 + r + 8 * hh;
        if (rowg < XBM && col < XNCLS)
          logits[(size_t)rowg * XNCLS + col] = a2[r] + bb;
      }
    }
    __syncthreads();
  }
}

// ---------------- host launcher ----------------
extern "C" void kernel_launch(void* const* d_in, const int* in_sizes, int n_in,
                              void* d_out, int out_size, void* d_ws, size_t ws_size,
                              hipStream_t stream) {
  (void)in_sizes; (void)n_in; (void)out_size; (void)ws_size;
  // d_in[0] = pts_feat_layers -- provably unused (scatter overwrites every gathered voxel)
  const float* pts_sample_feat = (const float*)d_in[1];
  const float* img_latent      = (const float*)d_in[2];
  const float* img_feat_layers = (const float*)d_in[3];
  const float* leaner_W = (const float*)d_in[4];
  const float* leaner_b = (const float*)d_in[5];
  const float* fcs1_W   = (const float*)d_in[6];
  const float* fcs1_b   = (const float*)d_in[7];
  const float* fcs2_W   = (const float*)d_in[8];
  const float* fcs2_b   = (const float*)d_in[9];
  const float* cls_W1   = (const float*)d_in[10];
  const float* cls_b1   = (const float*)d_in[11];
  const float* cls_W2   = (const float*)d_in[12];
  const float* cls_b2   = (const float*)d_in[13];
  const int* points_img = (const int*)d_in[14];
  const int* p2i        = (const int*)d_in[15];
  const int* samp_idx   = (const int*)d_in[16];
  const int* coors_inv  = (const int*)d_in[17];

  float* logits = (float*)d_out;                       // (BM, 20)
  float* out3d  = logits + (size_t)XBM * XNCLS;        // (BN, 512)

  // workspace carve (~65 MiB)
  char* ws = (char*)d_ws;
  int* vwin = (int*)ws; ws += sizeof(int) * XV;
  int* imgw = (int*)ws; ws += sizeof(int) * XBN;
  int* smpw = (int*)ws; ws += sizeof(int) * XBN;
  int* gidx = (int*)ws; ws += sizeof(int) * XBM;        // 2,560,000 B so far (16B aligned)
  _Float16* wtl = (_Float16*)ws; ws += (size_t)XNSC * 128 * 128 * 2;
  _Float16* wt1 = (_Float16*)ws; ws += (size_t)XNSC * 128 * 256 * 2;
  _Float16* wt2 = (_Float16*)ws; ws += (size_t)XNSC * 128 * 128 * 2;
  _Float16* cw1 = (_Float16*)ws; ws += (size_t)128 * 512 * 2;
  _Float16* cw2 = (_Float16*)ws; ws += (size_t)32 * 128 * 2;
  _Float16* seg = (_Float16*)ws;                        // (BM, 512) f16

  (void)hipMemsetAsync(vwin, 0xFF, sizeof(int) * XV, stream);
  (void)hipMemsetAsync(imgw, 0xFF, sizeof(int) * XBN, stream);
  (void)hipMemsetAsync(smpw, 0xFF, sizeof(int) * XBN, stream);

  // weight prep: transpose + convert to f16 once
  for (int s = 0; s < XNSC; ++s) {
    transpose_f16_kernel<<<(128 * 128 + 255) / 256, 256, 0, stream>>>(
        leaner_W + (size_t)s * 128 * 128, wtl + (size_t)s * 128 * 128, 128);
    transpose_f16_kernel<<<(256 * 128 + 255) / 256, 256, 0, stream>>>(
        fcs1_W + (size_t)s * 256 * 128, wt1 + (size_t)s * 128 * 256, 256);
    transpose_f16_kernel<<<(128 * 128 + 255) / 256, 256, 0, stream>>>(
        fcs2_W + (size_t)s * 128 * 128, wt2 + (size_t)s * 128 * 128, 128);
  }
  transpose_f16_kernel<<<(512 * 128 + 255) / 256, 256, 0, stream>>>(cls_W1, cw1, 512);
  pad_w2_kernel<<<(32 * 128 + 255) / 256, 256, 0, stream>>>(cls_W2, cw2);

  win_scatter_kernel<<<(XBM + 255) / 256, 256, 0, stream>>>(p2i, imgw, XBM, XM, XN);
  win_scatter_kernel<<<(XB * XS + 255) / 256, 256, 0, stream>>>(samp_idx, smpw, XB * XS, XS, XN);
  vox_winner_kernel<<<(XBN + 255) / 256, 256, 0, stream>>>(coors_inv, vwin);
  gidx_kernel<<<(XBM + 255) / 256, 256, 0, stream>>>(p2i, coors_inv, vwin, gidx);

  build_sp_kernel<<<XBN, XHID, 0, stream>>>(img_latent, pts_sample_feat, points_img,
                                            imgw, smpw, out3d);

  dim3 mlp_grid((XBM + MLP_RPS * MLP_STRIPS - 1) / (MLP_RPS * MLP_STRIPS), XNSC);
  fused_mlp_kernel<<<mlp_grid, 256, MLP_SMEM_BYTES, stream>>>(
      out3d, gidx, img_feat_layers, wtl, wt1, wt2,
      leaner_b, fcs1_b, fcs2_b, seg);

  classifier_kernel<<<(XBM + CLS_RPS * CLS_STRIPS - 1) / (CLS_RPS * CLS_STRIPS), 256,
                      CLS_SMEM_BYTES, stream>>>(seg, cw1, cw2, cls_b1, cls_b2, logits);
}
